// LinearAttention_50757923504215
// MI455X (gfx1250) — matrix operand order repaired and checked by reading, not compiled, not run on hardware
//
#include <hip/hip_runtime.h>
#include <hip/hip_bf16.h>
#include <cstdint>

// ---------------------------------------------------------------------------
// Problem constants (from reference)
// ---------------------------------------------------------------------------
#define BATCH 4
#define SEQ   4096
#define DIM   1024
#define HEADS 8
#define DH    128               // DIM / HEADS
#define MROWS (BATCH * SEQ)     // 16384 token rows

// ---------------------------------------------------------------------------
// WMMA vector types (CDNA5 gfx1250, wave32)
// ---------------------------------------------------------------------------
typedef __attribute__((ext_vector_type(16))) __bf16         v16bf;
typedef __attribute__((ext_vector_type(8)))  float          v8f;
typedef __attribute__((ext_vector_type(8)))  unsigned short v8us;

union BfPack16 {
    v8us  h[2];
    v16bf v;
};

// ---------------------------------------------------------------------------
// Explicit global-address-space accessors: guarantees global_load_b128 /
// global_store_* (avoids the flat_load_b128 the generic path produced).
// ---------------------------------------------------------------------------
#define GAS __attribute__((address_space(1)))

__device__ __forceinline__ v8us gload8(const unsigned short* p) {
    return *(const GAS v8us*)(unsigned long long)p;
}
__device__ __forceinline__ unsigned long long gload64(const unsigned short* p) {
    return *(const GAS unsigned long long*)(unsigned long long)p;
}
__device__ __forceinline__ void gstore_b16(unsigned short* p, unsigned short v) {
    *(GAS unsigned short*)(unsigned long long)p = v;
}
__device__ __forceinline__ void gstore_f32(float* p, float v) {
    *(GAS float*)(unsigned long long)p = v;
}

// ---------------------------------------------------------------------------
// bf16 <-> f32 helpers (round-to-nearest-even)
// ---------------------------------------------------------------------------
__device__ __forceinline__ unsigned short f2b(float f) {
    unsigned int u = __float_as_uint(f);
    unsigned int r = u + 0x7FFFu + ((u >> 16) & 1u);
    return (unsigned short)(r >> 16);
}
__device__ __forceinline__ float b2f(unsigned short h) {
    return __uint_as_float(((unsigned int)h) << 16);
}

// ---------------------------------------------------------------------------
// K1: fp32 -> bf16 conversion, 4 elements/thread (b128 load, b64 store)
// ---------------------------------------------------------------------------
__global__ void la_cvt_bf16(const float* __restrict__ src,
                            unsigned short* __restrict__ dst, size_t n4) {
    size_t i = (size_t)blockIdx.x * blockDim.x + threadIdx.x;
    const size_t stride = (size_t)gridDim.x * blockDim.x;
    for (; i < n4; i += stride) {
        const float4 f = reinterpret_cast<const float4*>(src)[i];
        unsigned long long p =
            (unsigned long long)f2b(f.x) |
            ((unsigned long long)f2b(f.y) << 16) |
            ((unsigned long long)f2b(f.z) << 32) |
            ((unsigned long long)f2b(f.w) << 48);
        reinterpret_cast<unsigned long long*>(dst)[i] = p;
    }
}

// ---------------------------------------------------------------------------
// K2: zero small fp32 accumulators
// ---------------------------------------------------------------------------
__global__ void la_zero_f32(float* __restrict__ p, int n) {
    int i = blockIdx.x * blockDim.x + threadIdx.x;
    if (i < n) p[i] = 0.0f;
}

// ---------------------------------------------------------------------------
// Shared WMMA GEMM body:  Y[m,n] = epi( sum_c A[m,c] * W[n,c] )
//   A: [M x K] row-major bf16, W: [N x K] row-major bf16 (y = A @ W^T).
//   Block = 256 threads = 8 waves. Wave w owns a 16(M) x 64(N) strip:
//   rows (blockIdx.x*8+w)*16, cols blockIdx.y*64 .. +63 (4 accumulators).
//   One A fetch feeds 4 WMMAs -> 2.5 b128 loads per v_wmma.
//
//   16-bit A 16x32 striping (contiguous k, same as B): lanes 0-15 hold row M with
//   K 0..15; lanes 16-31 hold row M with K 16..31.
//   B operand: column n of B = row n of W (contiguous); lanes 0-15 carry
//   K 0-15, lanes 16-31 carry K 16-31.
// ---------------------------------------------------------------------------
template <bool ACT, bool F32OUT>
__device__ __forceinline__ void la_gemm_body(const unsigned short* __restrict__ A,
                                             const unsigned short* __restrict__ W,
                                             unsigned short* __restrict__ Ybf,
                                             float* __restrict__ Yf,
                                             const float* __restrict__ bias,
                                             int K, int N) {
    const int lane   = threadIdx.x & 31;
    const int wave   = threadIdx.x >> 5;
    const int tileM  = (blockIdx.x * 8 + wave) * 16;
    const int tileN0 = blockIdx.y * 64;
    const int hi     = lane >> 4;            // 0: low half-lanes, 1: high
    const int rowA   = tileM + (lane & 15);

    const unsigned short* pa  = A + (size_t)rowA * K + hi * 16;
    const unsigned short* pb0 = W + (size_t)(tileN0 + (lane & 15)) * K + hi * 16;
    const unsigned short* pb1 = pb0 + (size_t)16 * K;
    const unsigned short* pb2 = pb0 + (size_t)32 * K;
    const unsigned short* pb3 = pb0 + (size_t)48 * K;

    v8f c0 = {}, c1 = {}, c2 = {}, c3 = {};
    for (int k0 = 0; k0 < K; k0 += 32) {
        BfPack16 ua, ub0, ub1, ub2, ub3;
        ua.h[0]  = gload8(pa);        // K +0..7   / +16..23
        ua.h[1]  = gload8(pa + 8);    // K +8..15  / +24..31
        ub0.h[0] = gload8(pb0);       // K +0..7   / +16..23
        ub0.h[1] = gload8(pb0 + 8);   // K +8..15  / +24..31
        ub1.h[0] = gload8(pb1);
        ub1.h[1] = gload8(pb1 + 8);
        ub2.h[0] = gload8(pb2);
        ub2.h[1] = gload8(pb2 + 8);
        ub3.h[0] = gload8(pb3);
        ub3.h[1] = gload8(pb3 + 8);
        __builtin_prefetch(pa  + 32, 0, 1);   // global/flat_prefetch_b8
        __builtin_prefetch(pb0 + 32, 0, 1);
        __builtin_prefetch(pb2 + 32, 0, 1);
        c0 = __builtin_amdgcn_wmma_f32_16x16x32_bf16(
                 false, ua.v, false, ub0.v, (short)0, c0, false, false);
        c1 = __builtin_amdgcn_wmma_f32_16x16x32_bf16(
                 false, ua.v, false, ub1.v, (short)0, c1, false, false);
        c2 = __builtin_amdgcn_wmma_f32_16x16x32_bf16(
                 false, ua.v, false, ub2.v, (short)0, c2, false, false);
        c3 = __builtin_amdgcn_wmma_f32_16x16x32_bf16(
                 false, ua.v, false, ub3.v, (short)0, c3, false, false);
        pa += 32; pb0 += 32; pb1 += 32; pb2 += 32; pb3 += 32;
    }

    // C/D layout: VGPR i -> M = tileM + hi*8 + i, N = tileN + (lane & 15)
    const int colb  = lane & 15;
    const int rbase = tileM + hi * 8;
    v8f accs[4] = {c0, c1, c2, c3};
#pragma unroll
    for (int j = 0; j < 4; ++j) {
        const int col = tileN0 + j * 16 + colb;
        float bn = 0.0f;
        if (F32OUT) bn = bias[col];
#pragma unroll
        for (int i = 0; i < 8; ++i) {
            float f = accs[j][i];
            if (ACT) f = (f > 0.0f) ? (f + 1.0f) : __expf(f);   // elu(x)+1
            if (F32OUT)
                gstore_f32(Yf + (size_t)(rbase + i) * N + col, f + bn);
            else
                gstore_b16(Ybf + (size_t)(rbase + i) * N + col, f2b(f));
        }
    }
}

// ---------------------------------------------------------------------------
// K3: fused Q/K/V projection (blockIdx.z selects weight/output/activation)
// ---------------------------------------------------------------------------
__global__ void la_wmma_qkv(const unsigned short* __restrict__ xb,
                            const unsigned short* __restrict__ wq,
                            const unsigned short* __restrict__ wk,
                            const unsigned short* __restrict__ wv,
                            unsigned short* __restrict__ qb,
                            unsigned short* __restrict__ kb,
                            unsigned short* __restrict__ vb,
                            int K, int N) {
    if (blockIdx.z == 0)
        la_gemm_body<true,  false>(xb, wq, qb, nullptr, nullptr, K, N);
    else if (blockIdx.z == 1)
        la_gemm_body<true,  false>(xb, wk, kb, nullptr, nullptr, K, N);
    else
        la_gemm_body<false, false>(xb, wv, vb, nullptr, nullptr, K, N);
}

// ---------------------------------------------------------------------------
// K4: final projection with bias, fp32 output
// ---------------------------------------------------------------------------
__global__ void la_wmma_out(const unsigned short* __restrict__ qs,
                            const unsigned short* __restrict__ wo,
                            const float* __restrict__ bias,
                            float* __restrict__ out,
                            int K, int N) {
    la_gemm_body<false, true>(qs, wo, nullptr, out, bias, K, N);
}

// ---------------------------------------------------------------------------
// K5: vsum[b,h,l] = sum_d v[b,l,h*DH+d].  One wave per (token, head);
//     lane handles 4 contiguous bf16 (one u64 load), butterfly reduce.
// ---------------------------------------------------------------------------
__global__ void la_vsum(const unsigned short* __restrict__ vb,
                        float* __restrict__ vsum) {
    const int gtid = blockIdx.x * blockDim.x + threadIdx.x;
    const int w    = gtid >> 5;          // wave id = (token, head)
    const int lane = threadIdx.x & 31;
    const int t = w >> 3;                // token index b*SEQ + l
    const int h = w & 7;

    const unsigned long long raw =
        gload64(vb + (size_t)t * DIM + h * DH + lane * 4);
    float s = b2f((unsigned short)(raw       )) +
              b2f((unsigned short)(raw >> 16)) +
              b2f((unsigned short)(raw >> 32)) +
              b2f((unsigned short)(raw >> 48));
#pragma unroll
    for (int o = 16; o >= 1; o >>= 1) s += __shfl_xor(s, o, 32);
    if (lane == 0) {
        const int b = t / SEQ, l = t % SEQ;
        vsum[((size_t)(b * HEADS + h)) * SEQ + l] = s;
    }
}

// ---------------------------------------------------------------------------
// K6: ksum[bh,d] = sum_l k[l,d];  s[bh,d] = sum_l k[l,d]*vsum[bh,l]
//     grid (B*H, LSPLIT), block = DH threads; fp32 atomics combine splits.
// ---------------------------------------------------------------------------
#define LSPLIT 16
__global__ void la_kstats(const unsigned short* __restrict__ kb,
                          const float* __restrict__ vsum,
                          float* __restrict__ ksum,
                          float* __restrict__ svec) {
    const int bh = blockIdx.x;           // 0..31
    const int d  = threadIdx.x;          // 0..127
    const int b  = bh >> 3, h = bh & 7;
    const int chunk = SEQ / LSPLIT;
    const int l0 = blockIdx.y * chunk;

    float ks = 0.0f, ss = 0.0f;
    for (int l = l0; l < l0 + chunk; ++l) {
        const float kv = b2f(kb[((size_t)(b * SEQ + l)) * DIM + h * DH + d]);
        const float vs = vsum[(size_t)bh * SEQ + l];
        ks += kv;
        ss += kv * vs;
    }
    atomicAdd(&ksum[bh * DH + d], ks);
    atomicAdd(&svec[bh * DH + d], ss);
}

// ---------------------------------------------------------------------------
// K7: qs[b,l,h,d] = q * s[b,h,d] / (q_row . ksum[b,h] + 1e-6)
//     One wave per (token, head); butterfly-reduced dot product.
// ---------------------------------------------------------------------------
__global__ void la_qscale(const unsigned short* __restrict__ qb,
                          const float* __restrict__ ksum,
                          const float* __restrict__ svec,
                          unsigned short* __restrict__ qs) {
    const int gtid = blockIdx.x * blockDim.x + threadIdx.x;
    const int w    = gtid >> 5;
    const int lane = threadIdx.x & 31;
    const int t = w >> 3;
    const int h = w & 7;
    const int b = t / SEQ;
    const int bh = b * HEADS + h;

    const size_t base = (size_t)t * DIM + h * DH + lane * 4;
    const unsigned long long raw = gload64(qb + base);
    const float q0 = b2f((unsigned short)(raw));
    const float q1 = b2f((unsigned short)(raw >> 16));
    const float q2 = b2f((unsigned short)(raw >> 32));
    const float q3 = b2f((unsigned short)(raw >> 48));

    const float* kspt = ksum + bh * DH + lane * 4;
    float dot = q0 * kspt[0] + q1 * kspt[1] + q2 * kspt[2] + q3 * kspt[3];
#pragma unroll
    for (int o = 16; o >= 1; o >>= 1) dot += __shfl_xor(dot, o, 32);

    const float z = 1.0f / (dot + 1e-6f);
    const float* spt = svec + bh * DH + lane * 4;
    const unsigned long long outp =
        (unsigned long long)f2b(q0 * spt[0] * z) |
        ((unsigned long long)f2b(q1 * spt[1] * z) << 16) |
        ((unsigned long long)f2b(q2 * spt[2] * z) << 32) |
        ((unsigned long long)f2b(q3 * spt[3] * z) << 48);
    *reinterpret_cast<unsigned long long*>(qs + base) = outp;
}

// ---------------------------------------------------------------------------
// Host orchestration
// ---------------------------------------------------------------------------
extern "C" void kernel_launch(void* const* d_in, const int* in_sizes, int n_in,
                              void* d_out, int out_size, void* d_ws, size_t ws_size,
                              hipStream_t stream) {
    const float* x  = (const float*)d_in[0];   // [B,L,DIM]
    const float* Wq = (const float*)d_in[1];   // [DIM,DIM]
    const float* Wk = (const float*)d_in[2];
    const float* Wv = (const float*)d_in[3];
    const float* Wo = (const float*)d_in[4];
    const float* bo = (const float*)d_in[5];   // [DIM]
    float* out = (float*)d_out;

    // Workspace layout (256B aligned slabs)
    char* ws = (char*)d_ws;
    size_t off = 0;
    auto slab = [&](size_t bytes) -> char* {
        char* p = ws + off;
        off += (bytes + 255) & ~(size_t)255;
        return p;
    };
    const size_t xbBytes = (size_t)MROWS * DIM * 2;   // 32 MB bf16
    const size_t wBytes  = (size_t)DIM * DIM * 2;     //  2 MB bf16
    unsigned short* xb  = (unsigned short*)slab(xbBytes);   // later reused as qs
    unsigned short* wqb = (unsigned short*)slab(wBytes);
    unsigned short* wkb = (unsigned short*)slab(wBytes);
    unsigned short* wvb = (unsigned short*)slab(wBytes);
    unsigned short* wob = (unsigned short*)slab(wBytes);
    unsigned short* qb  = (unsigned short*)slab(xbBytes);
    unsigned short* kb  = (unsigned short*)slab(xbBytes);
    unsigned short* vb  = (unsigned short*)slab(xbBytes);
    float* vsum = (float*)slab((size_t)BATCH * HEADS * SEQ * 4);   // 512 KB
    float* ksum = (float*)slab((size_t)BATCH * HEADS * DH * 4);    //  16 KB
    float* svec = (float*)slab((size_t)BATCH * HEADS * DH * 4);    //  16 KB
    (void)ws_size; (void)in_sizes; (void)n_in; (void)out_size;

    // 1) fp32 -> bf16 conversions (vectorized: 4 elems / thread)
    la_cvt_bf16<<<2048, 256, 0, stream>>>(x,  xb,  (size_t)MROWS * DIM / 4);
    la_cvt_bf16<<<512,  256, 0, stream>>>(Wq, wqb, (size_t)DIM * DIM / 4);
    la_cvt_bf16<<<512,  256, 0, stream>>>(Wk, wkb, (size_t)DIM * DIM / 4);
    la_cvt_bf16<<<512,  256, 0, stream>>>(Wv, wvb, (size_t)DIM * DIM / 4);
    la_cvt_bf16<<<512,  256, 0, stream>>>(Wo, wob, (size_t)DIM * DIM / 4);

    // 2) zero atomic accumulators (required per call: graph replays)
    la_zero_f32<<<(2 * BATCH * HEADS * DH + 255) / 256, 256, 0, stream>>>(
        ksum, 2 * BATCH * HEADS * DH);   // ksum and svec are contiguous slabs

    // 3) fused QKV projections: grid (M/128, N/64, 3), 8 waves/block,
    //    each wave computes a 16x64 strip (4 WMMA accumulators)
    dim3 qkvGrid(MROWS / 128, DIM / 64, 3);
    la_wmma_qkv<<<qkvGrid, 256, 0, stream>>>(xb, wqb, wkb, wvb,
                                             qb, kb, vb, DIM, DIM);

    // 4) v row-sums, then k statistics (ksum, s)
    const int nWaves = MROWS * HEADS;                   // 131072 waves
    la_vsum<<<(nWaves * 32) / 256, 256, 0, stream>>>(vb, vsum);
    la_kstats<<<dim3(BATCH * HEADS, LSPLIT), DH, 0, stream>>>(kb, vsum, ksum, svec);

    // 5) normalize + scale q -> qs (reuse xb slab; x no longer needed)
    unsigned short* qs = xb;
    la_qscale<<<(nWaves * 32) / 256, 256, 0, stream>>>(qb, ksum, svec, qs);

    // 6) output projection with bias (fp32 out)
    dim3 outGrid(MROWS / 128, DIM / 64);
    la_wmma_out<<<outGrid, 256, 0, stream>>>(qs, wob, bo, out, DIM, DIM);
}
